// GIN_49469433316049
// MI455X (gfx1250) — matrix operand order repaired and checked by reading, not compiled, not run on hardware
//
#include <hip/hip_runtime.h>
#include <hip/hip_bf16.h>

typedef __attribute__((ext_vector_type(16))) _Float16 v16h;
typedef __attribute__((ext_vector_type(8)))  float    v8f;

// ---------------------------------------------------------------------------
// WMMA fragment layouts per CDNA5 ISA 7.12.2 (wave32):
//   A 16x32 f16 : lane<16: row=lane,   halves 0..7 -> K=kk+0..7,  8..15 -> K=kk+16..23
//                 lane>=16: row=lane-16, halves 0..7 -> K=kk+8..15, 8..15 -> K=kk+24..31
//   B 32x16 f16 : col = lane&15, half i -> K = kk + 8*(lane>=16) + (i&7) + 16*(i>>3)
//   C/D 16x16 f32: col = lane&15, vgpr r -> row = r + 8*(lane>=16)
// Weights are pre-packed to f16 so each lane's 16 B-halves are contiguous:
//   Wp[((kb*nTiles + tile)*512) + lane*16 + i]
// ---------------------------------------------------------------------------

// Pack f32 weights [K, cols] into f16 B-fragment layout.
__global__ void k_packW(const float* __restrict__ W, _Float16* __restrict__ Wp,
                        int K, int cols) {
  const int tid = blockIdx.x * blockDim.x + threadIdx.x;
  const int total = K * cols;
  if (tid >= total) return;
  const int i    = tid & 15;
  const int lane = (tid >> 4) & 31;
  const int rest = tid >> 9;
  const int nT   = cols >> 4;
  const int tile = rest % nT;
  const int kb   = rest / nT;
  const int k    = kb * 32 + ((lane >> 4) << 3) + (i & 7) + ((i >> 3) << 4);
  const int col  = tile * 16 + (lane & 15);
  Wp[tid] = (_Float16)W[(size_t)k * cols + col];
}

// h = x[N,128] @ W[128,64] + b   (Wp packed, 4 col tiles)
__global__ void k_lin0(const float* __restrict__ x, const _Float16* __restrict__ Wp,
                       const float* __restrict__ b, float* __restrict__ h, int N) {
  const int wid = threadIdx.x >> 5, lane = threadIdx.x & 31;
  const int rowBase = (blockIdx.x * (blockDim.x >> 5) + wid) * 16;
  if (rowBase >= N) return;                      // wave-uniform: EXEC stays full
  const int hi = lane >> 4, l15 = lane & 15;
  const int row = rowBase + l15;
  v8f acc[4];
#pragma unroll
  for (int t = 0; t < 4; ++t) {
    float bv = b[t * 16 + l15];
#pragma unroll
    for (int i = 0; i < 8; ++i) acc[t][i] = bv;
  }
  for (int kk = 0; kk < 128; kk += 32) {
    const float4* ap = (const float4*)(x + (size_t)row * 128 + kk + hi * 8);
    const float4 a0 = ap[0], a1 = ap[1], a2 = ap[4], a3 = ap[5];
    v16h af;
    af[0] = (_Float16)a0.x;  af[1] = (_Float16)a0.y;  af[2] = (_Float16)a0.z;  af[3] = (_Float16)a0.w;
    af[4] = (_Float16)a1.x;  af[5] = (_Float16)a1.y;  af[6] = (_Float16)a1.z;  af[7] = (_Float16)a1.w;
    af[8] = (_Float16)a2.x;  af[9] = (_Float16)a2.y;  af[10] = (_Float16)a2.z; af[11] = (_Float16)a2.w;
    af[12] = (_Float16)a3.x; af[13] = (_Float16)a3.y; af[14] = (_Float16)a3.z; af[15] = (_Float16)a3.w;
#pragma unroll
    for (int t = 0; t < 4; ++t) {
      const v16h bf = *(const v16h*)(Wp + ((((kk >> 5) * 4 + t) << 9) + (lane << 4)));
      acc[t] = __builtin_amdgcn_wmma_f32_16x16x32_f16(false, af, false, bf,
                                                      (short)0, acc[t], false, false);
    }
  }
  const int rOff = hi * 8;
#pragma unroll
  for (int t = 0; t < 4; ++t)
#pragma unroll
    for (int r = 0; r < 8; ++r)
      h[(size_t)(rowBase + rOff + r) * 64 + t * 16 + l15] = acc[t][r];
}

// z = ((1+eps)*h + agg)[N,64] @ W[64,128] + b   (Wp packed, 8 col tiles)
__global__ void k_conv1(const float* __restrict__ h, const float* __restrict__ agg,
                        const float* __restrict__ epsP, const _Float16* __restrict__ Wp,
                        const float* __restrict__ b, float* __restrict__ z, int N) {
  const int wid = threadIdx.x >> 5, lane = threadIdx.x & 31;
  const int rowBase = (blockIdx.x * (blockDim.x >> 5) + wid) * 16;
  if (rowBase >= N) return;
  const int hi = lane >> 4, l15 = lane & 15;
  const int row = rowBase + l15;
  const float ce = 1.0f + epsP[0];
  v8f acc[8];
#pragma unroll
  for (int t = 0; t < 8; ++t) {
    float bv = b[t * 16 + l15];
#pragma unroll
    for (int i = 0; i < 8; ++i) acc[t][i] = bv;
  }
  for (int kk = 0; kk < 64; kk += 32) {
    const float4* hp = (const float4*)(h   + (size_t)row * 64 + kk + hi * 8);
    const float4* gp = (const float4*)(agg + (size_t)row * 64 + kk + hi * 8);
    float av[16];
    {
      const float4 h0 = hp[0], h1 = hp[1], h2 = hp[4], h3 = hp[5];
      const float4 g0 = gp[0], g1 = gp[1], g2 = gp[4], g3 = gp[5];
      av[0] = ce*h0.x+g0.x;  av[1] = ce*h0.y+g0.y;  av[2] = ce*h0.z+g0.z;  av[3] = ce*h0.w+g0.w;
      av[4] = ce*h1.x+g1.x;  av[5] = ce*h1.y+g1.y;  av[6] = ce*h1.z+g1.z;  av[7] = ce*h1.w+g1.w;
      av[8] = ce*h2.x+g2.x;  av[9] = ce*h2.y+g2.y;  av[10] = ce*h2.z+g2.z; av[11] = ce*h2.w+g2.w;
      av[12] = ce*h3.x+g3.x; av[13] = ce*h3.y+g3.y; av[14] = ce*h3.z+g3.z; av[15] = ce*h3.w+g3.w;
    }
    v16h af;
#pragma unroll
    for (int i = 0; i < 16; ++i) af[i] = (_Float16)av[i];
#pragma unroll
    for (int t = 0; t < 8; ++t) {
      const v16h bf = *(const v16h*)(Wp + ((((kk >> 5) * 8 + t) << 9) + (lane << 4)));
      acc[t] = __builtin_amdgcn_wmma_f32_16x16x32_f16(false, af, false, bf,
                                                      (short)0, acc[t], false, false);
    }
  }
  const int rOff = hi * 8;
#pragma unroll
  for (int t = 0; t < 8; ++t)
#pragma unroll
    for (int r = 0; r < 8; ++r)
      z[(size_t)(rowBase + rOff + r) * 128 + t * 16 + l15] = acc[t][r];
}

// hout = relu(z*scale+shift)[N,128] @ W[128,64] + b   (ss = {scale[128],shift[128]})
__global__ void k_conv2(const float* __restrict__ z, const float* __restrict__ ss,
                        const _Float16* __restrict__ Wp, const float* __restrict__ b,
                        float* __restrict__ hout, int N) {
  const int wid = threadIdx.x >> 5, lane = threadIdx.x & 31;
  const int rowBase = (blockIdx.x * (blockDim.x >> 5) + wid) * 16;
  if (rowBase >= N) return;
  const int hi = lane >> 4, l15 = lane & 15;
  const int row = rowBase + l15;
  v8f acc[4];
#pragma unroll
  for (int t = 0; t < 4; ++t) {
    float bv = b[t * 16 + l15];
#pragma unroll
    for (int i = 0; i < 8; ++i) acc[t][i] = bv;
  }
  for (int kk = 0; kk < 128; kk += 32) {
    const int kb = kk + hi * 8;
    const float4* zp  = (const float4*)(z  + (size_t)row * 128 + kb);
    const float4* scp = (const float4*)(ss + kb);
    const float4* shp = (const float4*)(ss + 128 + kb);
    v16h af;
    {
      const float4 z0 = zp[0], z1 = zp[1], z2 = zp[4], z3 = zp[5];
      const float4 c0 = scp[0], c1 = scp[1], c2 = scp[4], c3 = scp[5];
      const float4 s0 = shp[0], s1 = shp[1], s2 = shp[4], s3 = shp[5];
      af[0]  = (_Float16)fmaxf(z0.x*c0.x+s0.x, 0.f); af[1]  = (_Float16)fmaxf(z0.y*c0.y+s0.y, 0.f);
      af[2]  = (_Float16)fmaxf(z0.z*c0.z+s0.z, 0.f); af[3]  = (_Float16)fmaxf(z0.w*c0.w+s0.w, 0.f);
      af[4]  = (_Float16)fmaxf(z1.x*c1.x+s1.x, 0.f); af[5]  = (_Float16)fmaxf(z1.y*c1.y+s1.y, 0.f);
      af[6]  = (_Float16)fmaxf(z1.z*c1.z+s1.z, 0.f); af[7]  = (_Float16)fmaxf(z1.w*c1.w+s1.w, 0.f);
      af[8]  = (_Float16)fmaxf(z2.x*c2.x+s2.x, 0.f); af[9]  = (_Float16)fmaxf(z2.y*c2.y+s2.y, 0.f);
      af[10] = (_Float16)fmaxf(z2.z*c2.z+s2.z, 0.f); af[11] = (_Float16)fmaxf(z2.w*c2.w+s2.w, 0.f);
      af[12] = (_Float16)fmaxf(z3.x*c3.x+s3.x, 0.f); af[13] = (_Float16)fmaxf(z3.y*c3.y+s3.y, 0.f);
      af[14] = (_Float16)fmaxf(z3.z*c3.z+s3.z, 0.f); af[15] = (_Float16)fmaxf(z3.w*c3.w+s3.w, 0.f);
    }
#pragma unroll
    for (int t = 0; t < 4; ++t) {
      const v16h bf = *(const v16h*)(Wp + ((((kk >> 5) * 4 + t) << 9) + (lane << 4)));
      acc[t] = __builtin_amdgcn_wmma_f32_16x16x32_f16(false, af, false, bf,
                                                      (short)0, acc[t], false, false);
    }
  }
  const int rOff = hi * 8;
#pragma unroll
  for (int t = 0; t < 4; ++t)
#pragma unroll
    for (int r = 0; r < 8; ++r)
      hout[(size_t)(rowBase + rOff + r) * 64 + t * 16 + l15] = acc[t][r];
}

// agg[dst] += relu(h[src]); one wave per edge, 2 channels per lane (float2 gather)
__global__ void k_edge(const int* __restrict__ ei, const float* __restrict__ h,
                       float* __restrict__ agg, int E) {
  const int e = blockIdx.x * (blockDim.x >> 5) + (threadIdx.x >> 5);
  if (e >= E) return;
  const int lane = threadIdx.x & 31;
  const int src = ei[e], dst = ei[E + e];
  const float2 v = *(const float2*)(h + (size_t)src * 64 + lane * 2);
  float* d = agg + (size_t)dst * 64 + lane * 2;
  unsafeAtomicAdd(d,     fmaxf(v.x, 0.0f));
  unsafeAtomicAdd(d + 1, fmaxf(v.y, 0.0f));
}

// per-column sum & sumsq -> stats[{0..cols-1}=sum, {cols..}=sumsq]
__global__ void k_bnstats(const float* __restrict__ a, float* __restrict__ stats,
                          int N, int cols) {
  const int col = threadIdx.x;                    // blockDim.x == cols
  const int rowsPerBlock = (N + gridDim.x - 1) / gridDim.x;
  int r0 = blockIdx.x * rowsPerBlock;
  int r1 = r0 + rowsPerBlock; if (r1 > N) r1 = N;
  float s = 0.0f, s2 = 0.0f;
  for (int r = r0; r < r1; ++r) {
    float v = a[(size_t)r * cols + col];
    s += v; s2 += v * v;
  }
  unsafeAtomicAdd(&stats[col], s);
  unsafeAtomicAdd(&stats[cols + col], s2);
}

__global__ void k_bnfin(const float* __restrict__ stats, const float* __restrict__ gamma,
                        const float* __restrict__ beta, float* __restrict__ ss,
                        int N, int cols) {
  const int c = threadIdx.x;
  const float inv = 1.0f / (float)N;
  float mean = stats[c] * inv;
  float var  = stats[cols + c] * inv - mean * mean;
  float sc   = gamma[c] * rsqrtf(var + 1e-5f);
  ss[c] = sc;
  ss[cols + c] = beta[c] - mean * sc;
}

__global__ void k_bnapply(float* __restrict__ h, const float* __restrict__ ss, int total) {
  int i = blockIdx.x * blockDim.x + threadIdx.x;
  if (i >= total) return;
  int c = i & 63;
  h[i] = h[i] * ss[c] + ss[64 + c];
}

// gate = h @ gw + gb; per-graph running max via ordered-uint atomicMax
__global__ void k_gate(const float* __restrict__ h, const float* __restrict__ gw,
                       const float* __restrict__ gb, const int* __restrict__ batch,
                       float* __restrict__ gate, unsigned* __restrict__ gmaxkey, int N) {
  const int n = blockIdx.x * (blockDim.x >> 5) + (threadIdx.x >> 5);
  if (n >= N) return;
  const int lane = threadIdx.x & 31;
  float d = h[(size_t)n * 64 + lane] * gw[lane] +
            h[(size_t)n * 64 + 32 + lane] * gw[32 + lane];
#pragma unroll
  for (int m = 16; m; m >>= 1) d += __shfl_xor(d, m, 32);
  if (lane == 0) {
    float g = d + gb[0];
    gate[n] = g;
    unsigned u = __float_as_uint(g);
    unsigned key = (u & 0x80000000u) ? ~u : (u | 0x80000000u);
    atomicMax(&gmaxkey[batch[n]], key);
  }
}

__global__ void k_expw(const float* __restrict__ gate, const unsigned* __restrict__ gmaxkey,
                       const int* __restrict__ batch, float* __restrict__ w,
                       float* __restrict__ wsum, int N) {
  int n = blockIdx.x * blockDim.x + threadIdx.x;
  if (n >= N) return;
  int g = batch[n];
  unsigned key = gmaxkey[g];
  unsigned u = (key & 0x80000000u) ? (key ^ 0x80000000u) : ~key;
  float wv = __expf(gate[n] - __uint_as_float(u));
  w[n] = wv;
  unsafeAtomicAdd(&wsum[g], wv);
}

__global__ void k_pool(const float* __restrict__ h, const float* __restrict__ w,
                       const float* __restrict__ wsum, const int* __restrict__ batch,
                       float* __restrict__ pooled, int N) {
  const int n = blockIdx.x * (blockDim.x >> 5) + (threadIdx.x >> 5);
  if (n >= N) return;
  const int lane = threadIdx.x & 31;
  const int g = batch[n];
  const float alpha = w[n] / wsum[g];
  unsafeAtomicAdd(&pooled[(size_t)g * 64 + lane],      alpha * h[(size_t)n * 64 + lane]);
  unsafeAtomicAdd(&pooled[(size_t)g * 64 + 32 + lane], alpha * h[(size_t)n * 64 + 32 + lane]);
}

// out[g] = relu(pooled @ pred_W + pred_b) @ cls_W + cls_b  (G=128, single block)
__global__ void k_head(const float* __restrict__ pooled, const float* __restrict__ pW,
                       const float* __restrict__ pb, const float* __restrict__ cW,
                       const float* __restrict__ cb, float* __restrict__ out, int G) {
  __shared__ float sW[64 * 64];
  __shared__ float sC[64 * 2];
  for (int i = threadIdx.x; i < 64 * 64; i += blockDim.x) sW[i] = pW[i];
  for (int i = threadIdx.x; i < 128;     i += blockDim.x) sC[i] = cW[i];
  __syncthreads();
  const int g = blockIdx.x * blockDim.x + threadIdx.x;
  if (g >= G) return;
  float row[64];
#pragma unroll
  for (int k = 0; k < 64; ++k) row[k] = pooled[(size_t)g * 64 + k];
  float o0 = cb[0], o1 = cb[1];
  for (int j = 0; j < 64; ++j) {
    float a = pb[j];
#pragma unroll
    for (int k = 0; k < 64; ++k) a += row[k] * sW[k * 64 + j];
    a = fmaxf(a, 0.0f);
    o0 += a * sC[j * 2 + 0];
    o1 += a * sC[j * 2 + 1];
  }
  out[g * 2 + 0] = o0;
  out[g * 2 + 1] = o1;
}

extern "C" void kernel_launch(void* const* d_in, const int* in_sizes, int n_in,
                              void* d_out, int out_size, void* d_ws, size_t ws_size,
                              hipStream_t stream) {
  const float* x      = (const float*)d_in[0];
  const float* lin0_W = (const float*)d_in[1];
  const float* lin0_b = (const float*)d_in[2];
  const float* epsArr = (const float*)d_in[3];
  const float* W1     = (const float*)d_in[4];
  const float* b1     = (const float*)d_in[5];
  const float* g1     = (const float*)d_in[6];
  const float* be1    = (const float*)d_in[7];
  const float* W2     = (const float*)d_in[8];
  const float* b2     = (const float*)d_in[9];
  const float* gbn    = (const float*)d_in[10];
  const float* bbn    = (const float*)d_in[11];
  const float* gate_W = (const float*)d_in[12];
  const float* gate_b = (const float*)d_in[13];
  const float* pred_W = (const float*)d_in[14];
  const float* pred_b = (const float*)d_in[15];
  const float* cls_W  = (const float*)d_in[16];
  const float* cls_b  = (const float*)d_in[17];
  const int* edge_index = (const int*)d_in[18];
  const int* batch      = (const int*)d_in[19];

  const int N = in_sizes[19];
  const int E = in_sizes[18] / 2;
  const int L = 3, H = 64, G = 128;
  const int WSZ = 128 * 64;                      // elements per packed weight (8192)

  float* ws = (float*)d_ws;
  size_t off = 0;
  float* hA      = ws + off; off += (size_t)N * H;
  float* agg     = ws + off; off += (size_t)N * H;
  float* z       = ws + off; off += (size_t)N * 2 * H;
  float* stats   = ws + off; off += 256;
  float* ss      = ws + off; off += 256;
  float* gate    = ws + off; off += N;
  float* wv      = ws + off; off += N;
  float* wsum    = ws + off; off += G;
  unsigned* gmaxkey = (unsigned*)(ws + off); off += G;
  float* pooled  = ws + off; off += (size_t)G * H;
  _Float16* wp0 = (_Float16*)(ws + off);         // packed lin0_W
  _Float16* wp1 = wp0 + WSZ;                     // packed W1[0..2]
  _Float16* wp2 = wp1 + 3 * (size_t)WSZ;         // packed W2[0..2]

  const int WPB = 8, TPB = WPB * 32;
  const int gemmBlocks = (N / 16 + WPB - 1) / WPB;
  const int edgeBlocks = (E + WPB - 1) / WPB;
  const int nodeBlocks = (N + WPB - 1) / WPB;
  const int packBlocks = (WSZ + 255) / 256;

  // Pack all weights into f16 B-fragment layout (tiny, once per launch).
  k_packW<<<packBlocks, 256, 0, stream>>>(lin0_W, wp0, 128, 64);
  for (int l = 0; l < L; ++l) {
    k_packW<<<packBlocks, 256, 0, stream>>>(W1 + (size_t)l * WSZ, wp1 + (size_t)l * WSZ, 64, 128);
    k_packW<<<packBlocks, 256, 0, stream>>>(W2 + (size_t)l * WSZ, wp2 + (size_t)l * WSZ, 128, 64);
  }

  k_lin0<<<gemmBlocks, TPB, 0, stream>>>(x, wp0, lin0_b, hA, N);

  for (int l = 0; l < L; ++l) {
    hipMemsetAsync(agg, 0, (size_t)N * H * sizeof(float), stream);
    k_edge<<<edgeBlocks, TPB, 0, stream>>>(edge_index, hA, agg, E);
    k_conv1<<<gemmBlocks, TPB, 0, stream>>>(hA, agg, epsArr + l,
        wp1 + (size_t)l * WSZ, b1 + (size_t)l * 2 * H, z, N);
    hipMemsetAsync(stats, 0, 256 * sizeof(float), stream);
    k_bnstats<<<512, 128, 0, stream>>>(z, stats, N, 128);
    k_bnfin<<<1, 128, 0, stream>>>(stats, g1 + (size_t)l * 2 * H,
                                   be1 + (size_t)l * 2 * H, ss, N, 128);
    k_conv2<<<gemmBlocks, TPB, 0, stream>>>(z, ss,
        wp2 + (size_t)l * WSZ, b2 + (size_t)l * H, hA, N);
    if (l >= 1) {
      hipMemsetAsync(stats, 0, 256 * sizeof(float), stream);
      k_bnstats<<<512, 64, 0, stream>>>(hA, stats, N, 64);
      k_bnfin<<<1, 64, 0, stream>>>(stats, gbn + (size_t)(l - 1) * H,
                                    bbn + (size_t)(l - 1) * H, ss, N, 64);
      const int tot = N * H;
      k_bnapply<<<(tot + 255) / 256, 256, 0, stream>>>(hA, ss, tot);
    }
  }

  hipMemsetAsync(gmaxkey, 0, G * sizeof(unsigned), stream);
  hipMemsetAsync(wsum, 0, G * sizeof(float), stream);
  hipMemsetAsync(pooled, 0, (size_t)G * H * sizeof(float), stream);
  k_gate<<<nodeBlocks, TPB, 0, stream>>>(hA, gate_W, gate_b, batch, gate, gmaxkey, N);
  k_expw<<<(N + 255) / 256, 256, 0, stream>>>(gate, gmaxkey, batch, wv, wsum, N);
  k_pool<<<nodeBlocks, TPB, 0, stream>>>(hA, wv, wsum, batch, pooled, N);
  k_head<<<1, 128, 0, stream>>>(pooled, pred_W, pred_b, cls_W, cls_b, (float*)d_out, G);
}
